// BiLSTMCRF_37907381354725
// MI455X (gfx1250) — compile-verified, matrix operand order checked by reading, not run on hardware
//
#include <hip/hip_runtime.h>
#include <hip/hip_bf16.h>

// ---------------- problem constants ----------------
#define TT   8192      // tokens
#define VV   512       // vocab
#define NWW  2048      // words
#define EE   1024      // emb size
#define HH   768       // hidden
#define G4H  3072      // 4*H
#define H2   1536      // 2*H
#define FCH  384       // H/2
#define LL   13        // labels

typedef __attribute__((ext_vector_type(8)))  __bf16 v8bf;
typedef __attribute__((ext_vector_type(16))) __bf16 v16bf;
typedef __attribute__((ext_vector_type(8)))  float  v8f;

// ---------------- small helper kernels ----------------

__global__ void f32_to_bf16_kernel(const float* __restrict__ in, __bf16* __restrict__ out, int n) {
    int i = blockIdx.x * blockDim.x + threadIdx.x;
    if (i < n) out[i] = (__bf16)in[i];
}

// emb_bf16[t][e] = bf16(embedding[tok[t]][e]); 4 elements per thread
__global__ void gather_embed_kernel(const int* __restrict__ tok,
                                    const float* __restrict__ emb,
                                    __bf16* __restrict__ out) {
    int idx = (blockIdx.x * blockDim.x + threadIdx.x) * 4;
    int t = idx >> 10;          // / EE
    int e = idx & (EE - 1);
    int v = tok[t];
    const float* src = emb + (size_t)v * EE + e;
    out[idx + 0] = (__bf16)src[0];
    out[idx + 1] = (__bf16)src[1];
    out[idx + 2] = (__bf16)src[2];
    out[idx + 3] = (__bf16)src[3];
}

// W_hh [4H][H] -> W_hh^T [H][4H] (coalesced recurrent matvec reads)
__global__ void transpose_whh_kernel(const float* __restrict__ in, float* __restrict__ out) {
    int idx = blockIdx.x * blockDim.x + threadIdx.x;
    if (idx >= G4H * HH) return;
    int j = idx / HH;   // gate row
    int k = idx % HH;   // hidden dim
    out[(size_t)k * G4H + j] = in[idx];
}

// ---------------- WMMA bf16 GEMM: C[M,N] = A[M,K] * Bw[N,K]^T + bias (opt relu) ----
// block = 256 threads (8 waves); block tile 128(M) x 128(N); K stage 64 (2 chunks).
// waves arranged 4(M) x 2(N); each wave owns a 32x64 subtile => acc[2][4].
// Per stage per wave: 12 fragment loads feed 16 WMMA (1.5 b128 per WMMA).

#define LDK 72   // padded LDS row (64 data + 8 pad bf16 = 144B, 16B aligned, bank-skewed)

__device__ __forceinline__ v16bf load_frag_lds(const __bf16* rowp, int k0) {
    // per CDNA5 ISA 16-bit A/B layout: two 8-element runs, 16 apart in K
    v8bf lo = *(const v8bf*)(rowp + k0);
    v8bf hi = *(const v8bf*)(rowp + k0 + 16);
    return __builtin_shufflevector(lo, hi, 0,1,2,3,4,5,6,7,8,9,10,11,12,13,14,15);
}

__global__ void __launch_bounds__(256)
wmma_gemm_bias_kernel(const __bf16* __restrict__ A,
                      const __bf16* __restrict__ Bw,
                      const float* __restrict__ bias,
                      float* __restrict__ C,
                      int M, int N, int K, int relu) {
    __shared__ __bf16 As[128][LDK];
    __shared__ __bf16 Bs[128][LDK];

    const int m0 = blockIdx.y * 128;
    const int n0 = blockIdx.x * 128;

    const int lane = threadIdx.x & 31;
    const int wave = threadIdx.x >> 5;
    const int wm = (wave & 3) * 32;       // wave M offset in block tile
    const int wn = (wave >> 2) * 64;      // wave N offset in block tile
    const int row = lane & 15;            // A row / B col within 16
    const int k0  = ((lane >> 4) & 1) * 8;

    // staging indices: thread copies 32 contiguous elems of one row half
    const int ldr   = threadIdx.x >> 1;        // 0..127
    const int ldoff = (threadIdx.x & 1) * 32;  // 0 / 32

    v8f acc[2][4] = {};

    for (int kt = 0; kt < K; kt += 64) {
        // ---- stage A tile: 128 x 64 bf16 ----
        {
            const __bf16* src = A + (size_t)(m0 + ldr) * K + kt + ldoff;
            __bf16* dst = &As[ldr][ldoff];
            *(v8bf*)(dst + 0)  = *(const v8bf*)(src + 0);
            *(v8bf*)(dst + 8)  = *(const v8bf*)(src + 8);
            *(v8bf*)(dst + 16) = *(const v8bf*)(src + 16);
            *(v8bf*)(dst + 24) = *(const v8bf*)(src + 24);
            if (kt + 64 < K) __builtin_prefetch(src + 64, 0, 1);   // global_prefetch_b8
        }
        // ---- stage B tile: 128 x 64 bf16 ----
        {
            const __bf16* src = Bw + (size_t)(n0 + ldr) * K + kt + ldoff;
            __bf16* dst = &Bs[ldr][ldoff];
            *(v8bf*)(dst + 0)  = *(const v8bf*)(src + 0);
            *(v8bf*)(dst + 8)  = *(const v8bf*)(src + 8);
            *(v8bf*)(dst + 16) = *(const v8bf*)(src + 16);
            *(v8bf*)(dst + 24) = *(const v8bf*)(src + 24);
            if (kt + 64 < K) __builtin_prefetch(src + 64, 0, 1);
        }
        __syncthreads();

        #pragma unroll
        for (int kc = 0; kc < 64; kc += 32) {
            v16bf a0 = load_frag_lds(&As[wm + row][kc],      k0);
            v16bf a1 = load_frag_lds(&As[wm + 16 + row][kc], k0);
            v16bf b0 = load_frag_lds(&Bs[wn + row][kc],      k0);
            v16bf b1 = load_frag_lds(&Bs[wn + 16 + row][kc], k0);
            v16bf b2 = load_frag_lds(&Bs[wn + 32 + row][kc], k0);
            v16bf b3 = load_frag_lds(&Bs[wn + 48 + row][kc], k0);

            acc[0][0] = __builtin_amdgcn_wmma_f32_16x16x32_bf16(false, a0, false, b0, (short)0, acc[0][0], false, false);
            acc[0][1] = __builtin_amdgcn_wmma_f32_16x16x32_bf16(false, a0, false, b1, (short)0, acc[0][1], false, false);
            acc[0][2] = __builtin_amdgcn_wmma_f32_16x16x32_bf16(false, a0, false, b2, (short)0, acc[0][2], false, false);
            acc[0][3] = __builtin_amdgcn_wmma_f32_16x16x32_bf16(false, a0, false, b3, (short)0, acc[0][3], false, false);
            acc[1][0] = __builtin_amdgcn_wmma_f32_16x16x32_bf16(false, a1, false, b0, (short)0, acc[1][0], false, false);
            acc[1][1] = __builtin_amdgcn_wmma_f32_16x16x32_bf16(false, a1, false, b1, (short)0, acc[1][1], false, false);
            acc[1][2] = __builtin_amdgcn_wmma_f32_16x16x32_bf16(false, a1, false, b2, (short)0, acc[1][2], false, false);
            acc[1][3] = __builtin_amdgcn_wmma_f32_16x16x32_bf16(false, a1, false, b3, (short)0, acc[1][3], false, false);
        }
        __syncthreads();
    }

    // ---- epilogue: D layout — lane half selects rows r / r+8; col = lane&15 ----
    const int rhalf = ((lane >> 4) & 1) * 8;
    for (int i = 0; i < 2; ++i) {
        for (int j = 0; j < 4; ++j) {
            int col = n0 + wn + j * 16 + (lane & 15);
            float bv = bias[col];
            int rbase = m0 + wm + i * 16 + rhalf;
            for (int r = 0; r < 8; ++r) {
                float v = acc[i][j][r] + bv;
                if (relu) v = fmaxf(v, 0.0f);
                C[(size_t)(rbase + r) * N + col] = v;
            }
        }
    }
}

// ---------------- sequential BiLSTM scan ----------------
// grid = 2 (dir), block = 1024. W^T in [H][4H] layout for coalescing; h,c,gates in LDS.

__device__ __forceinline__ float sigm(float x) { return 1.0f / (1.0f + expf(-x)); }

__global__ void lstm_kernel(const float* __restrict__ xp_f, const float* __restrict__ xp_b,
                            const float* __restrict__ WtF,  const float* __restrict__ WtB,
                            float* __restrict__ hs) {
    __shared__ float h_s[HH];
    __shared__ float c_s[HH];
    __shared__ float gates[G4H];

    const int dir = blockIdx.x;
    const float* xp = dir ? xp_b : xp_f;
    const float* Wt = dir ? WtB : WtF;     // [HH][G4H]
    const int tid = threadIdx.x;

    for (int j = tid; j < HH; j += 1024) { h_s[j] = 0.0f; c_s[j] = 0.0f; }
    __syncthreads();

    for (int t = 0; t < TT; ++t) {
        const int idx = dir ? (TT - 1 - t) : t;
        // recurrent matvec: thread owns gate rows tid, tid+1024, tid+2048
        float a0 = 0.0f, a1 = 0.0f, a2 = 0.0f;
        for (int k = 0; k < HH; ++k) {
            float hv = h_s[k];
            const float* wr = Wt + (size_t)k * G4H;
            a0 = fmaf(hv, wr[tid],        a0);
            a1 = fmaf(hv, wr[tid + 1024], a1);
            a2 = fmaf(hv, wr[tid + 2048], a2);
        }
        const float* xr = xp + (size_t)idx * G4H;
        gates[tid]        = xr[tid]        + a0;
        gates[tid + 1024] = xr[tid + 1024] + a1;
        gates[tid + 2048] = xr[tid + 2048] + a2;
        __syncthreads();

        if (tid < HH) {
            float ig = sigm(gates[tid]);
            float fg = sigm(gates[HH + tid]);
            float gg = tanhf(gates[2 * HH + tid]);
            float og = sigm(gates[3 * HH + tid]);
            float c  = fg * c_s[tid] + ig * gg;
            float h  = og * tanhf(c);
            c_s[tid] = c;
            h_s[tid] = h;
            hs[(size_t)idx * H2 + dir * HH + tid] = h;
        }
        __syncthreads();
    }
}

// ---------------- ragged mean pool (segment_ids sorted) ----------------

__device__ __forceinline__ int lower_bound_dev(const int* a, int n, int v) {
    int lo = 0, hi = n;
    while (lo < hi) { int mid = (lo + hi) >> 1; if (a[mid] < v) lo = mid + 1; else hi = mid; }
    return lo;
}

__global__ void pool_kernel(const int* __restrict__ seg, const float* __restrict__ hs,
                            __bf16* __restrict__ pooled) {
    const int w = blockIdx.x;
    const int lo = lower_bound_dev(seg, TT, w);
    const int hi = lower_bound_dev(seg, TT, w + 1);
    const float inv = 1.0f / fmaxf((float)(hi - lo), 1.0f);
    for (int d = threadIdx.x; d < H2; d += blockDim.x) {
        float s = 0.0f;
        for (int t = lo; t < hi; ++t) s += hs[(size_t)t * H2 + d];
        pooled[(size_t)w * H2 + d] = (__bf16)(s * inv);
    }
}

// ---------------- fc2 (N=13, VALU) + weighted CE loss ----------------

__global__ void fc2_kernel(const float* __restrict__ z, const float* __restrict__ w,
                           const float* __restrict__ b, float* __restrict__ logits) {
    int idx = blockIdx.x * blockDim.x + threadIdx.x;
    if (idx >= NWW * LL) return;
    int wd = idx / LL, l = idx % LL;
    const float* zr = z + (size_t)wd * FCH;
    const float* wr = w + (size_t)l * FCH;
    float s = b[l];
    for (int k = 0; k < FCH; ++k) s = fmaf(zr[k], wr[k], s);
    logits[idx] = s;
}

__global__ void loss_kernel(const float* __restrict__ logits, const int* __restrict__ gold,
                            const float* __restrict__ cw, float* __restrict__ out) {
    __shared__ float s_wn[256];
    __shared__ float s_w[256];
    float wn = 0.0f, ww = 0.0f;
    for (int wd = threadIdx.x; wd < NWW; wd += 256) {
        const float* l = logits + wd * LL;
        float m = l[0];
        for (int i = 1; i < LL; ++i) m = fmaxf(m, l[i]);
        float s = 0.0f;
        for (int i = 0; i < LL; ++i) s += expf(l[i] - m);
        float lse = m + logf(s);
        int g = gold[wd];
        float w = cw[g];
        wn += w * (lse - l[g]);
        ww += w;
    }
    s_wn[threadIdx.x] = wn; s_w[threadIdx.x] = ww;
    __syncthreads();
    for (int st = 128; st > 0; st >>= 1) {
        if (threadIdx.x < st) { s_wn[threadIdx.x] += s_wn[threadIdx.x + st]; s_w[threadIdx.x] += s_w[threadIdx.x + st]; }
        __syncthreads();
    }
    if (threadIdx.x == 0) out[0] = s_wn[0] / s_w[0];
}

// ---------------- host launcher ----------------

extern "C" void kernel_launch(void* const* d_in, const int* in_sizes, int n_in,
                              void* d_out, int out_size, void* d_ws, size_t ws_size,
                              hipStream_t stream) {
    const int*   inp_tok     = (const int*)  d_in[0];
    const int*   segment_ids = (const int*)  d_in[1];
    const int*   gold_lab    = (const int*)  d_in[2];
    const float* embedding   = (const float*)d_in[3];
    const float* W_ih_f      = (const float*)d_in[4];
    const float* W_hh_f      = (const float*)d_in[5];
    const float* b_f         = (const float*)d_in[6];
    const float* W_ih_b      = (const float*)d_in[7];
    const float* W_hh_b      = (const float*)d_in[8];
    const float* b_b         = (const float*)d_in[9];
    const float* fc1_w       = (const float*)d_in[10];
    const float* fc1_b       = (const float*)d_in[11];
    const float* fc2_w       = (const float*)d_in[12];
    const float* fc2_b       = (const float*)d_in[13];
    const float* class_wts   = (const float*)d_in[14];
    float* out = (float*)d_out;

    // workspace carve-out (256B aligned)
    char* ws = (char*)d_ws;
    auto alloc = [&](size_t bytes) -> char* {
        char* p = ws; ws += (bytes + 255) & ~(size_t)255; return p;
    };
    __bf16* emb_bf   = (__bf16*)alloc((size_t)TT * EE * 2);
    __bf16* wihf_bf  = (__bf16*)alloc((size_t)G4H * EE * 2);
    __bf16* wihb_bf  = (__bf16*)alloc((size_t)G4H * EE * 2);
    __bf16* fc1w_bf  = (__bf16*)alloc((size_t)FCH * H2 * 2);
    float*  xp_f     = (float*) alloc((size_t)TT * G4H * 4);
    float*  xp_b     = (float*) alloc((size_t)TT * G4H * 4);
    float*  whhfT    = (float*) alloc((size_t)HH * G4H * 4);
    float*  whhbT    = (float*) alloc((size_t)HH * G4H * 4);
    float*  hs       = (float*) alloc((size_t)TT * H2 * 4);
    __bf16* pooled   = (__bf16*)alloc((size_t)NWW * H2 * 2);
    float*  z        = (float*) alloc((size_t)NWW * FCH * 4);
    float*  logits   = (float*) alloc((size_t)NWW * LL * 4);

    // 1) precision conversions + gather
    {
        int n = G4H * EE;
        f32_to_bf16_kernel<<<(n + 255) / 256, 256, 0, stream>>>(W_ih_f, wihf_bf, n);
        f32_to_bf16_kernel<<<(n + 255) / 256, 256, 0, stream>>>(W_ih_b, wihb_bf, n);
        int n2 = FCH * H2;
        f32_to_bf16_kernel<<<(n2 + 255) / 256, 256, 0, stream>>>(fc1_w, fc1w_bf, n2);
    }
    gather_embed_kernel<<<(TT * EE / 4) / 256, 256, 0, stream>>>(inp_tok, embedding, emb_bf);

    // 2) input projections via WMMA bf16 GEMM (the ~100 GFLOP bulk)
    {
        dim3 grid(G4H / 128, TT / 128);
        wmma_gemm_bias_kernel<<<grid, 256, 0, stream>>>(emb_bf, wihf_bf, b_f, xp_f, TT, G4H, EE, 0);
        wmma_gemm_bias_kernel<<<grid, 256, 0, stream>>>(emb_bf, wihb_bf, b_b, xp_b, TT, G4H, EE, 0);
    }

    // 3) transpose W_hh for coalesced recurrent matvec; run both directions
    {
        int n = G4H * HH;
        transpose_whh_kernel<<<(n + 255) / 256, 256, 0, stream>>>(W_hh_f, whhfT);
        transpose_whh_kernel<<<(n + 255) / 256, 256, 0, stream>>>(W_hh_b, whhbT);
    }
    lstm_kernel<<<2, 1024, 0, stream>>>(xp_f, xp_b, whhfT, whhbT, hs);

    // 4) ragged mean pool -> bf16
    pool_kernel<<<NWW, 256, 0, stream>>>(segment_ids, hs, pooled);

    // 5) fc1 via WMMA (relu)
    {
        dim3 grid(FCH / 128, NWW / 128);
        wmma_gemm_bias_kernel<<<grid, 256, 0, stream>>>(pooled, fc1w_bf, fc1_b, z, NWW, FCH, H2, 1);
    }

    // 6) fc2 + weighted cross-entropy
    fc2_kernel<<<(NWW * LL + 255) / 256, 256, 0, stream>>>(z, fc2_w, fc2_b, logits);
    loss_kernel<<<1, 256, 0, stream>>>(logits, gold_lab, class_wts, out);
    (void)in_sizes; (void)n_in; (void)out_size; (void)ws_size;
}
